// BEiTAttention_46926812676894
// MI455X (gfx1250) — matrix operand order repaired and checked by reading, not compiled, not run on hardware
//
#include <hip/hip_runtime.h>
#include <hip/hip_bf16.h>

typedef __attribute__((ext_vector_type(16))) __bf16 v16bf;
typedef __attribute__((ext_vector_type(8)))  float  v8f;

#define EMBED 768
#define HEADS 12
#define HD    64
#define NTOK  1025
#define BATCH 16
#define MROWS (BATCH * NTOK)       /* 16400 (1025 tiles of 16) */
#define QKVN  (3 * EMBED)          /* 2304 */
#define QTILES ((NTOK + 15) / 16)  /* 65 */
#define KBLKS  ((NTOK + 31) / 32)  /* 33 */

static __device__ __forceinline__ unsigned short f32_to_bf16(float f) {
  unsigned u = __float_as_uint(f);
  u += 0x7FFFu + ((u >> 16) & 1u);        // round-to-nearest-even
  return (unsigned short)(u >> 16);
}
static __device__ __forceinline__ __bf16 bf_from_us(unsigned short s) {
  union { unsigned short u; __bf16 b; } v; v.u = s; return v.b;
}
static __device__ __forceinline__ int imin(int a, int b) { return a < b ? a : b; }

static __device__ __forceinline__ float hmax16(float v) {
  v = fmaxf(v, __shfl_xor(v, 1, 32));
  v = fmaxf(v, __shfl_xor(v, 2, 32));
  v = fmaxf(v, __shfl_xor(v, 4, 32));
  v = fmaxf(v, __shfl_xor(v, 8, 32));
  return v;
}
static __device__ __forceinline__ float hsum16(float v) {
  v += __shfl_xor(v, 1, 32);
  v += __shfl_xor(v, 2, 32);
  v += __shfl_xor(v, 4, 32);
  v += __shfl_xor(v, 8, 32);
  return v;
}

// ---------------- fp32 -> bf16 conversion ----------------
__global__ __launch_bounds__(256) void cvt_f32_bf16(const float* __restrict__ src,
                                                    unsigned short* __restrict__ dst, int n) {
  int i = blockIdx.x * 256 + threadIdx.x;
  if (i < n) dst[i] = f32_to_bf16(src[i]);
}

// ------------- relative-position bias gather (H,N,N) f32 -------------
__global__ __launch_bounds__(256) void bias_gather(const float* __restrict__ table,
                                                   const int* __restrict__ index,
                                                   float* __restrict__ biasbuf) {
  int p = blockIdx.x * 256 + threadIdx.x;
  if (p >= NTOK * NTOK) return;
  int idx = index[p];
  #pragma unroll
  for (int h = 0; h < HEADS; ++h)
    biasbuf[(size_t)h * NTOK * NTOK + p] = table[idx * HEADS + h];
}

// ---------------- QKV projection GEMM (bf16 WMMA, f32 acc) ----------------
// Each wave computes a 16(M) x 64(N) tile of qkv = x @ Wqkv^T, then scatters
// q (scaled by 1/8, +q_bias), k, v (+v_bias) as bf16 into (B,H,N,d) buffers.
__global__ __launch_bounds__(256) void gemm_qkv(const unsigned short* __restrict__ xbf,
                                                const unsigned short* __restrict__ wbf,
                                                const float* __restrict__ qbias,
                                                const float* __restrict__ vbias,
                                                unsigned short* __restrict__ qb,
                                                unsigned short* __restrict__ kb,
                                                unsigned short* __restrict__ vb) {
  const int NT = QKVN / 64; // 36 column tiles of 64
  int lane = threadIdx.x & 31;
  int wave = blockIdx.x * 8 + (threadIdx.x >> 5);
  if (wave >= (MROWS / 16) * NT) return;
  int n64 = wave % NT, mt = wave / NT;
  int half = lane >> 4, ln = lane & 15;
  int m0 = mt * 16;

  v8f acc[4] = {};
  for (int k0 = 0; k0 < EMBED; k0 += 32) {
    v16bf a;
    const unsigned short* ab = xbf + (size_t)(m0 + ln) * EMBED + k0;
    #pragma unroll
    for (int s = 0; s < 16; ++s) {
      int k = (half << 4) + s;
      a[s] = bf_from_us(ab[k]);
    }
    #pragma unroll
    for (int t = 0; t < 4; ++t) {
      const unsigned short* bb =
          wbf + (size_t)(n64 * 64 + t * 16 + ln) * EMBED + k0 + (half << 4);
      v16bf bfr;
      #pragma unroll
      for (int s = 0; s < 16; ++s) bfr[s] = bf_from_us(bb[s]);
      acc[t] = __builtin_amdgcn_wmma_f32_16x16x32_bf16(false, a, false, bfr,
                                                       (short)0, acc[t], false, false);
    }
  }

  #pragma unroll
  for (int t = 0; t < 4; ++t) {
    int col = n64 * 64 + t * 16 + ln;
    float bias = (col < EMBED) ? qbias[col]
               : (col < 2 * EMBED) ? 0.0f : vbias[col - 2 * EMBED];
    int which = col / EMBED;
    int cc = col - which * EMBED;
    int h = cc >> 6, d = cc & 63;
    unsigned short* dst = (which == 0) ? qb : (which == 1) ? kb : vb;
    #pragma unroll
    for (int r = 0; r < 8; ++r) {
      int mg = m0 + r + (half << 3);
      float v = acc[t][r] + bias;
      if (which == 0) v *= 0.125f;          // fold hd^-0.5 into q
      int bi = mg / NTOK;
      int nq = mg - bi * NTOK;
      dst[(((size_t)bi * HEADS + h) * NTOK + nq) * HD + d] = f32_to_bf16(v);
    }
  }
}

// ---------------- flash attention (bf16 WMMA, f32 softmax) ----------------
__global__ __launch_bounds__(256) void attn_flash(const unsigned short* __restrict__ qb,
                                                  const unsigned short* __restrict__ kb,
                                                  const unsigned short* __restrict__ vb,
                                                  const float* __restrict__ biasbuf,
                                                  unsigned short* __restrict__ aout) {
  __shared__ unsigned short pbuf[8][16 * 32];   // per-wave P tile (bf16)
  int lane = threadIdx.x & 31;
  int widx = threadIdx.x >> 5;
  int wave = blockIdx.x * 8 + widx;             // grid sized exactly: all active
  int qt = wave % QTILES;
  int bh = wave / QTILES;                       // b*HEADS + h
  int h  = bh % HEADS;
  int b  = bh / HEADS;
  int half = lane >> 4, ln = lane & 15;
  int q0 = qt * 16;

  const unsigned short* qbase = qb + (size_t)bh * NTOK * HD;
  const unsigned short* kbase = kb + (size_t)bh * NTOK * HD;
  const unsigned short* vbase = vb + (size_t)bh * NTOK * HD;
  const float* bbase = biasbuf + (size_t)h * NTOK * NTOK;

  // Q A-fragments: row = q0 + ln (clamped), K split 0..31 / 32..63
  int qm = imin(q0 + ln, NTOK - 1);
  v16bf qa0, qa1;
  #pragma unroll
  for (int s = 0; s < 16; ++s) {
    int k = (half << 4) + s;
    qa0[s] = bf_from_us(qbase[(size_t)qm * HD + k]);
    qa1[s] = bf_from_us(qbase[(size_t)qm * HD + k + 32]);
  }

  v8f o[4] = {};
  float mrow[8], lrow[8];
  #pragma unroll
  for (int r = 0; r < 8; ++r) { mrow[r] = -1e30f; lrow[r] = 0.0f; }

  for (int kblk = 0; kblk < KBLKS; ++kblk) {
    int kq = kblk * 32;
    // S = q . k^T for two 16-key subtiles
    v8f s0 = {}, s1 = {};
    {
      int key = imin(kq + ln, NTOK - 1);
      v16bf b0, b1;
      #pragma unroll
      for (int s = 0; s < 16; ++s) {
        int kd = (half << 4) + s;
        b0[s] = bf_from_us(kbase[(size_t)key * HD + kd]);
        b1[s] = bf_from_us(kbase[(size_t)key * HD + kd + 32]);
      }
      s0 = __builtin_amdgcn_wmma_f32_16x16x32_bf16(false, qa0, false, b0, (short)0, s0, false, false);
      s0 = __builtin_amdgcn_wmma_f32_16x16x32_bf16(false, qa1, false, b1, (short)0, s0, false, false);
    }
    {
      int key = imin(kq + 16 + ln, NTOK - 1);
      v16bf b0, b1;
      #pragma unroll
      for (int s = 0; s < 16; ++s) {
        int kd = (half << 4) + s;
        b0[s] = bf_from_us(kbase[(size_t)key * HD + kd]);
        b1[s] = bf_from_us(kbase[(size_t)key * HD + kd + 32]);
      }
      s1 = __builtin_amdgcn_wmma_f32_16x16x32_bf16(false, qa0, false, b0, (short)0, s1, false, false);
      s1 = __builtin_amdgcn_wmma_f32_16x16x32_bf16(false, qa1, false, b1, (short)0, s1, false, false);
    }

    int key0 = kq + ln, key1 = kq + 16 + ln;
    int kc0 = imin(key0, NTOK - 1), kc1 = imin(key1, NTOK - 1);
    #pragma unroll
    for (int r = 0; r < 8; ++r) {
      int qr = imin(q0 + r + (half << 3), NTOK - 1);
      float a0 = s0[r] + bbase[(size_t)qr * NTOK + kc0];
      float a1 = s1[r] + bbase[(size_t)qr * NTOK + kc1];
      if (key0 >= NTOK) a0 = -1e30f;
      if (key1 >= NTOK) a1 = -1e30f;
      float mnew = fmaxf(mrow[r], hmax16(fmaxf(a0, a1)));
      float alpha = __expf(mrow[r] - mnew);
      mrow[r] = mnew;
      float p0 = __expf(a0 - mnew);
      float p1 = __expf(a1 - mnew);
      lrow[r] = lrow[r] * alpha + hsum16(p0 + p1);
      o[0][r] *= alpha; o[1][r] *= alpha; o[2][r] *= alpha; o[3][r] *= alpha;
      int m = r + (half << 3);
      pbuf[widx][m * 32 + ln]      = f32_to_bf16(p0);
      pbuf[widx][m * 32 + 16 + ln] = f32_to_bf16(p1);
    }
    __syncthreads();
    // reload P in A layout (transpose across lanes via LDS)
    v16bf pa;
    #pragma unroll
    for (int s = 0; s < 16; ++s) {
      int k = (half << 4) + s;
      pa[s] = bf_from_us(pbuf[widx][ln * 32 + k]);
    }
    __syncthreads();
    // O += P . V  (4 column tiles of d)
    #pragma unroll
    for (int t = 0; t < 4; ++t) {
      v16bf vf;
      #pragma unroll
      for (int s = 0; s < 16; ++s) {
        int key = imin(kq + (half << 4) + s, NTOK - 1);
        vf[s] = bf_from_us(vbase[(size_t)key * HD + t * 16 + ln]);
      }
      o[t] = __builtin_amdgcn_wmma_f32_16x16x32_bf16(false, pa, false, vf, (short)0, o[t], false, false);
    }
  }

  #pragma unroll
  for (int r = 0; r < 8; ++r) {
    int qr = q0 + r + (half << 3);
    if (qr >= NTOK) continue;
    float inv = lrow[r] > 0.0f ? 1.0f / lrow[r] : 0.0f;
    #pragma unroll
    for (int t = 0; t < 4; ++t)
      aout[((size_t)b * NTOK + qr) * EMBED + h * HD + t * 16 + ln] =
          f32_to_bf16(o[t][r] * inv);
  }
}

// ---------------- output projection GEMM (bf16 WMMA, f32 out) ----------------
__global__ __launch_bounds__(256) void gemm_proj(const unsigned short* __restrict__ abf,
                                                 const unsigned short* __restrict__ wbf,
                                                 const float* __restrict__ pbias,
                                                 float* __restrict__ out) {
  const int NT = EMBED / 64; // 12
  int lane = threadIdx.x & 31;
  int wave = blockIdx.x * 8 + (threadIdx.x >> 5);
  if (wave >= (MROWS / 16) * NT) return;
  int n64 = wave % NT, mt = wave / NT;
  int half = lane >> 4, ln = lane & 15;
  int m0 = mt * 16;

  v8f acc[4] = {};
  for (int k0 = 0; k0 < EMBED; k0 += 32) {
    v16bf a;
    const unsigned short* ab = abf + (size_t)(m0 + ln) * EMBED + k0;
    #pragma unroll
    for (int s = 0; s < 16; ++s) {
      int k = (half << 4) + s;
      a[s] = bf_from_us(ab[k]);
    }
    #pragma unroll
    for (int t = 0; t < 4; ++t) {
      const unsigned short* bb =
          wbf + (size_t)(n64 * 64 + t * 16 + ln) * EMBED + k0 + (half << 4);
      v16bf bfr;
      #pragma unroll
      for (int s = 0; s < 16; ++s) bfr[s] = bf_from_us(bb[s]);
      acc[t] = __builtin_amdgcn_wmma_f32_16x16x32_bf16(false, a, false, bfr,
                                                       (short)0, acc[t], false, false);
    }
  }
  #pragma unroll
  for (int t = 0; t < 4; ++t) {
    int col = n64 * 64 + t * 16 + ln;
    float bias = pbias[col];
    #pragma unroll
    for (int r = 0; r < 8; ++r) {
      int mg = m0 + r + (half << 3);
      out[(size_t)mg * EMBED + col] = acc[t][r] + bias;
    }
  }
}

extern "C" void kernel_launch(void* const* d_in, const int* in_sizes, int n_in,
                              void* d_out, int out_size, void* d_ws, size_t ws_size,
                              hipStream_t stream) {
  const float* x      = (const float*)d_in[0];
  const float* qkvw   = (const float*)d_in[1];
  const float* q_bias = (const float*)d_in[2];
  const float* v_bias = (const float*)d_in[3];
  const float* table  = (const float*)d_in[4];
  const float* projw  = (const float*)d_in[5];
  const float* p_bias = (const float*)d_in[6];
  const int*   relidx = (const int*)d_in[7];
  float* out = (float*)d_out;

  char* ws = (char*)d_ws;
  size_t off = 0;
  auto alloc = [&](size_t bytes) -> char* {
    char* p = ws + off;
    off = (off + bytes + 255) & ~(size_t)255;
    return p;
  };
  unsigned short* xbf   = (unsigned short*)alloc((size_t)MROWS * EMBED * 2);
  unsigned short* wqkv  = (unsigned short*)alloc((size_t)QKVN * EMBED * 2);
  unsigned short* wproj = (unsigned short*)alloc((size_t)EMBED * EMBED * 2);
  unsigned short* qb    = (unsigned short*)alloc((size_t)BATCH * HEADS * NTOK * HD * 2);
  unsigned short* kb    = (unsigned short*)alloc((size_t)BATCH * HEADS * NTOK * HD * 2);
  unsigned short* vb    = (unsigned short*)alloc((size_t)BATCH * HEADS * NTOK * HD * 2);
  float*          bbuf  = (float*)alloc((size_t)HEADS * NTOK * NTOK * 4);
  unsigned short* abuf  = (unsigned short*)alloc((size_t)MROWS * EMBED * 2);

  int nx = MROWS * EMBED, nwq = QKVN * EMBED, nwp = EMBED * EMBED;
  cvt_f32_bf16<<<(nx  + 255) / 256, 256, 0, stream>>>(x,     xbf,   nx);
  cvt_f32_bf16<<<(nwq + 255) / 256, 256, 0, stream>>>(qkvw,  wqkv,  nwq);
  cvt_f32_bf16<<<(nwp + 255) / 256, 256, 0, stream>>>(projw, wproj, nwp);
  bias_gather<<<(NTOK * NTOK + 255) / 256, 256, 0, stream>>>(table, relidx, bbuf);

  int waves_qkv = (MROWS / 16) * (QKVN / 64);          // 36900
  gemm_qkv<<<(waves_qkv + 7) / 8, 256, 0, stream>>>(xbf, wqkv, q_bias, v_bias, qb, kb, vb);

  int waves_att = BATCH * HEADS * QTILES;              // 12480, divisible by 8
  attn_flash<<<waves_att / 8, 256, 0, stream>>>(qb, kb, vb, bbuf, abuf);

  int waves_prj = (MROWS / 16) * (EMBED / 64);         // 12300
  gemm_proj<<<(waves_prj + 7) / 8, 256, 0, stream>>>(abuf, wproj, p_bias, out);
}
